// MultiHeadAttention_38628935860567
// MI455X (gfx1250) — matrix operand order repaired and checked by reading, not compiled, not run on hardware
//
#include <hip/hip_runtime.h>
#include <hip/hip_bf16.h>

// ---------------------------------------------------------------------------
// MI455X (gfx1250) multi-head causal attention, bf16 WMMA pipeline.
// Loads are batched per K-step (one clause + one wait, then back-to-back
// WMMAs) so the matrix pipe is not latency-bound on each B fragment.
// ---------------------------------------------------------------------------

typedef __attribute__((ext_vector_type(16))) __bf16 v16bf;
typedef __attribute__((ext_vector_type(8)))  float  v8f;
typedef __attribute__((ext_vector_type(4)))  int    i4v;   // trivial 16B vector (union-safe)

#define BB  4
#define TT  1024
#define EE  1024
#define HH  16
#define DH  64
#define HDH (HH * DH)   // 1024

// ---- bf16 helpers (round-to-nearest-even) ---------------------------------
__device__ __forceinline__ unsigned short f32_to_bf16u(float f) {
    unsigned int u = __float_as_uint(f);
    u += 0x7FFFu + ((u >> 16) & 1u);
    return (unsigned short)(u >> 16);
}

// ---- 16-lane butterfly reductions via ds_swizzle (xor masks stay in-half) --
__device__ __forceinline__ float redmax16(float v) {
    v = fmaxf(v, __int_as_float(__builtin_amdgcn_ds_swizzle(__float_as_int(v), 0x041F)));
    v = fmaxf(v, __int_as_float(__builtin_amdgcn_ds_swizzle(__float_as_int(v), 0x081F)));
    v = fmaxf(v, __int_as_float(__builtin_amdgcn_ds_swizzle(__float_as_int(v), 0x101F)));
    v = fmaxf(v, __int_as_float(__builtin_amdgcn_ds_swizzle(__float_as_int(v), 0x201F)));
    return v;
}
__device__ __forceinline__ float redsum16(float v) {
    v += __int_as_float(__builtin_amdgcn_ds_swizzle(__float_as_int(v), 0x041F));
    v += __int_as_float(__builtin_amdgcn_ds_swizzle(__float_as_int(v), 0x081F));
    v += __int_as_float(__builtin_amdgcn_ds_swizzle(__float_as_int(v), 0x101F));
    v += __int_as_float(__builtin_amdgcn_ds_swizzle(__float_as_int(v), 0x201F));
    return v;
}

__device__ __forceinline__ v8f wmma_bf16(v16bf a, v16bf b, v8f c) {
    return __builtin_amdgcn_wmma_f32_16x16x32_bf16(false, a, false, b, (short)0, c,
                                                   false, false);
}

union AFrag { v16bf v; i4v x[2]; };

// ---------------------------------------------------------------------------
// Kernel 0a: fp32 -> bf16 elementwise convert (x, Wo)
// ---------------------------------------------------------------------------
__global__ __launch_bounds__(256) void cvt_bf16_kernel(const float* __restrict__ src,
                                                       unsigned short* __restrict__ dst,
                                                       int n) {
    int i = blockIdx.x * blockDim.x + threadIdx.x;
    if (i < n) dst[i] = f32_to_bf16u(src[i]);
}

// ---------------------------------------------------------------------------
// Kernel 0b: W (H,E,DH) fp32 -> WT (H*DH, E) bf16  (K-contiguous B operand)
// ---------------------------------------------------------------------------
__global__ __launch_bounds__(256) void transpose_w_kernel(const float* __restrict__ src,
                                                          unsigned short* __restrict__ dst) {
    int idx = blockIdx.x * blockDim.x + threadIdx.x;        // over H*DH*E
    if (idx >= HH * DH * EE) return;
    int h = idx / (DH * EE);
    int r = idx - h * (DH * EE);
    int d = r / EE;
    int k = r - d * EE;
    dst[idx] = f32_to_bf16u(src[((size_t)h * EE + k) * DH + d]);
}

// ---------------------------------------------------------------------------
// Kernel 1: QKV projection.  One wave -> 16 (t rows) x 64 (dh) for one head,
// producing q,k in [B,H,T,DH] and v transposed in [B,H,DH,T] (all bf16).
// Per 32-wide K step: load A + all 12 B fragments (one wait), then issue
// 12 WMMAs back-to-back.
// ---------------------------------------------------------------------------
__global__ __launch_bounds__(256) void qkv_kernel(const unsigned short* __restrict__ xb,
                                                  const unsigned short* __restrict__ WqT,
                                                  const unsigned short* __restrict__ WkT,
                                                  const unsigned short* __restrict__ WvT,
                                                  unsigned short* __restrict__ qb,
                                                  unsigned short* __restrict__ kb,
                                                  unsigned short* __restrict__ vTb) {
    const int wave  = blockIdx.x * (blockDim.x >> 5) + (threadIdx.x >> 5);
    const int lane  = threadIdx.x & 31;
    const int lo    = lane & 15;
    const int hi    = lane >> 4;
    const int mtile = wave >> 4;      // 0..255 over B*T/16
    const int h     = wave & 15;
    const int m0    = mtile * 16;
    const int b     = m0 / TT;
    const int t0    = m0 - b * TT;

    v8f accq[4] = {}, acck[4] = {}, accv[4] = {};

    const unsigned short* arow_base = xb + (size_t)(m0 + lo) * EE + hi * 16;
    const size_t nrow = (size_t)(h * DH + lo) * EE + hi * 16;   // B row base (nt adds 16*EE)

#pragma unroll 2
    for (int k0 = 0; k0 < EE; k0 += 32) {
        // ---- load phase: A + 12 B fragments, single dependency point
        AFrag A;
        const unsigned short* arow = arow_base + k0;
        A.x[0] = *(const i4v*)(arow);
        A.x[1] = *(const i4v*)(arow + 8);
        __builtin_prefetch(arow + 64, 0, 1);   // global_prefetch_b8

        v16bf bq[4], bk[4], bv[4];
#pragma unroll
        for (int nt = 0; nt < 4; ++nt) {
            const size_t off = nrow + (size_t)nt * 16 * EE + k0;
            bq[nt] = *(const v16bf*)(WqT + off);
            bk[nt] = *(const v16bf*)(WkT + off);
            bv[nt] = *(const v16bf*)(WvT + off);
        }
        // ---- compute phase: 12 back-to-back WMMAs, A reused by all
#pragma unroll
        for (int nt = 0; nt < 4; ++nt) accq[nt] = wmma_bf16(A.v, bq[nt], accq[nt]);
#pragma unroll
        for (int nt = 0; nt < 4; ++nt) acck[nt] = wmma_bf16(A.v, bk[nt], acck[nt]);
#pragma unroll
        for (int nt = 0; nt < 4; ++nt) accv[nt] = wmma_bf16(A.v, bv[nt], accv[nt]);
    }

    const size_t bh = (size_t)(b * HH + h);
#pragma unroll
    for (int nt = 0; nt < 4; ++nt) {
#pragma unroll
        for (int r = 0; r < 8; ++r) {
            const int t  = t0 + r + hi * 8;
            const int dh = nt * 16 + lo;
            const size_t qi = (bh * TT + t) * DH + dh;
            qb[qi] = f32_to_bf16u(accq[nt][r]);
            kb[qi] = f32_to_bf16u(acck[nt][r]);
            const size_t vi = (bh * DH + dh) * TT + t;
            vTb[vi] = f32_to_bf16u(accv[nt][r]);
        }
    }
}

// ---------------------------------------------------------------------------
// Kernel 2: causal flash attention. One wave per (b,h,16-query tile).
// Per 32-key chunk: 4 WMMAs (QK^T) + online softmax (ds_swizzle row
// reductions) + LDS relayout of P (C-layout -> A-layout) + 4 WMMAs (PV).
// Control flow is wave-uniform, so EXEC is all-ones for every WMMA.
// ---------------------------------------------------------------------------
#define ATTN_WAVES 4
__global__ __launch_bounds__(ATTN_WAVES * 32) void attn_kernel(
        const unsigned short* __restrict__ qb,
        const unsigned short* __restrict__ kb,
        const unsigned short* __restrict__ vTb,
        unsigned short* __restrict__ attout) {       // [B*T][H*DH]
    __shared__ __align__(16) unsigned short pstage[ATTN_WAVES][16 * 32];

    const int wslot = threadIdx.x >> 5;
    const int lane  = threadIdx.x & 31;
    const int lo    = lane & 15;
    const int hi    = lane >> 4;
    const int wave  = blockIdx.x * ATTN_WAVES + wslot;
    const int mt    = wave & 63;         // 64 query tiles per (b,h)
    const int bh    = wave >> 6;         // 0..63
    const int m0    = mt * 16;

    const unsigned short* qh = qb  + (size_t)bh * TT * DH;
    const unsigned short* kh = kb  + (size_t)bh * TT * DH;
    const unsigned short* vh = vTb + (size_t)bh * DH * TT;

    // Preload the 16x64 Q tile as two A fragments (K = 0..31 and 32..63).
    AFrag Aq[2];
#pragma unroll
    for (int kk = 0; kk < 2; ++kk) {
        const unsigned short* p = qh + (size_t)(m0 + lo) * DH + kk * 32 + hi * 16;
        Aq[kk].x[0] = *(const i4v*)(p);
        Aq[kk].x[1] = *(const i4v*)(p + 8);
    }

    const float scale = 0.125f;          // 1/sqrt(64)
    const float NEG   = -1.0e30f;
    float mrow[8], lrow[8];
#pragma unroll
    for (int r = 0; r < 8; ++r) { mrow[r] = NEG; lrow[r] = 0.0f; }
    v8f acc[4] = {};

    unsigned short* myp = pstage[wslot];

    for (int kv0 = 0; kv0 <= m0 + 15; kv0 += 32) {
        // ---- load all 4 K fragments (2 key sub-tiles x 2 dh halves) + 4 V
        v16bf bK[2][2], bV[4];
#pragma unroll
        for (int st = 0; st < 2; ++st) {
            const size_t krow = (size_t)(kv0 + st * 16 + lo) * DH + hi * 16;
            bK[st][0] = *(const v16bf*)(kh + krow);
            bK[st][1] = *(const v16bf*)(kh + krow + 32);
        }
#pragma unroll
        for (int nt = 0; nt < 4; ++nt)
            bV[nt] = *(const v16bf*)(vh + (size_t)(nt * 16 + lo) * TT + kv0 + hi * 16);

        // ---- scores: two 16x16 tiles, 4 back-to-back WMMAs
        v8f sc[2];
#pragma unroll
        for (int st = 0; st < 2; ++st) {
            v8f s = {};
            s = wmma_bf16(Aq[0].v, bK[st][0], s);
            s = wmma_bf16(Aq[1].v, bK[st][1], s);
            sc[st] = s;
        }

        // ---- online softmax (rows r+8*hi live in register r, col = lo)
#pragma unroll
        for (int r = 0; r < 8; ++r) {
            const int qrow = m0 + r + hi * 8;
            float vmax = NEG;
#pragma unroll
            for (int st = 0; st < 2; ++st) {
                const int key = kv0 + st * 16 + lo;
                float sv = sc[st][r] * scale;
                sv = (key <= qrow) ? sv : NEG;
                sc[st][r] = sv;
                vmax = fmaxf(vmax, sv);
            }
            vmax = redmax16(vmax);
            const float mnew = fmaxf(mrow[r], vmax);
            const float corr = __expf(mrow[r] - mnew);
            mrow[r] = mnew;
            float rsum = 0.0f;
#pragma unroll
            for (int st = 0; st < 2; ++st) {
                float p = __expf(sc[st][r] - mnew);
                sc[st][r] = p;
                rsum += p;
            }
            rsum = redsum16(rsum);
            lrow[r] = lrow[r] * corr + rsum;
#pragma unroll
            for (int nt = 0; nt < 4; ++nt) acc[nt][r] *= corr;
        }

        // ---- relayout P through wave-private LDS: C-layout -> A-layout
#pragma unroll
        for (int r = 0; r < 8; ++r) {
            const int row = r + hi * 8;
#pragma unroll
            for (int st = 0; st < 2; ++st)
                myp[row * 32 + st * 16 + lo] = f32_to_bf16u(sc[st][r]);
        }
        AFrag Ap;
        {
            const unsigned short* prow = myp + lo * 32 + hi * 16;
            Ap.x[0] = *(const i4v*)(prow);
            Ap.x[1] = *(const i4v*)(prow + 8);
        }

        // ---- O += P @ V : 4 back-to-back WMMAs (V fragments preloaded)
#pragma unroll
        for (int nt = 0; nt < 4; ++nt)
            acc[nt] = wmma_bf16(Ap.v, bV[nt], acc[nt]);
    }

    // ---- normalize + store to [B*T][H*DH] bf16
    const int b = bh / HH;
    const int h = bh - b * HH;
#pragma unroll
    for (int nt = 0; nt < 4; ++nt) {
#pragma unroll
        for (int r = 0; r < 8; ++r) {
            const float o = acc[nt][r] / lrow[r];
            const int t = m0 + r + hi * 8;
            attout[(size_t)(b * TT + t) * HDH + h * DH + nt * 16 + lo] = f32_to_bf16u(o);
        }
    }
}

// ---------------------------------------------------------------------------
// Kernel 3: output projection  out = attout @ Wo^T + bo  (fp32 out).
// One wave -> 16 x 64 output tile; batched loads per K-step.
// ---------------------------------------------------------------------------
__global__ __launch_bounds__(256) void outproj_kernel(const unsigned short* __restrict__ ao,
                                                      const unsigned short* __restrict__ Wob,
                                                      const float* __restrict__ bo,
                                                      float* __restrict__ out) {
    const int wave  = blockIdx.x * (blockDim.x >> 5) + (threadIdx.x >> 5);
    const int lane  = threadIdx.x & 31;
    const int lo    = lane & 15;
    const int hi    = lane >> 4;
    const int mtile = wave >> 4;        // 256 m-tiles
    const int n64   = wave & 15;        // 16 n-tiles of 64
    const int m0    = mtile * 16;
    const int n0    = n64 * 64;

    v8f acc[4] = {};
    const unsigned short* arow_base = ao + (size_t)(m0 + lo) * HDH + hi * 16;
    const size_t nrow = (size_t)(n0 + lo) * HDH + hi * 16;

#pragma unroll 2
    for (int k0 = 0; k0 < HDH; k0 += 32) {
        AFrag A;
        const unsigned short* arow = arow_base + k0;
        A.x[0] = *(const i4v*)(arow);
        A.x[1] = *(const i4v*)(arow + 8);
        __builtin_prefetch(arow + 64, 0, 1);

        v16bf bW[4];
#pragma unroll
        for (int nt = 0; nt < 4; ++nt)
            bW[nt] = *(const v16bf*)(Wob + nrow + (size_t)nt * 16 * HDH + k0);
#pragma unroll
        for (int nt = 0; nt < 4; ++nt)
            acc[nt] = wmma_bf16(A.v, bW[nt], acc[nt]);
    }
#pragma unroll
    for (int nt = 0; nt < 4; ++nt) {
        const int n = n0 + nt * 16 + lo;
        const float bias = bo[n];
#pragma unroll
        for (int r = 0; r < 8; ++r)
            out[(size_t)(m0 + r + hi * 8) * EE + n] = acc[nt][r] + bias;
    }
}

// ---------------------------------------------------------------------------
extern "C" void kernel_launch(void* const* d_in, const int* in_sizes, int n_in,
                              void* d_out, int out_size, void* d_ws, size_t ws_size,
                              hipStream_t stream) {
    const float* x  = (const float*)d_in[0];
    const float* Wq = (const float*)d_in[1];
    const float* Wk = (const float*)d_in[2];
    const float* Wv = (const float*)d_in[3];
    const float* Wo = (const float*)d_in[4];
    const float* bo = (const float*)d_in[5];
    float* out = (float*)d_out;

    unsigned short* ws = (unsigned short*)d_ws;
    size_t off = 0;
    unsigned short* xb  = ws + off; off += (size_t)BB * TT * EE;   // x bf16      (8 MB)
    unsigned short* WqT = ws + off; off += (size_t)HDH * EE;       // Wq^T bf16   (2 MB)
    unsigned short* WkT = ws + off; off += (size_t)HDH * EE;
    unsigned short* WvT = ws + off; off += (size_t)HDH * EE;
    unsigned short* Wob = ws + off; off += (size_t)EE * HDH;       // Wo bf16
    unsigned short* qb  = ws + off; off += (size_t)BB * HH * TT * DH;
    unsigned short* kb  = ws + off; off += (size_t)BB * HH * TT * DH;
    unsigned short* vTb = ws + off; off += (size_t)BB * HH * DH * TT;
    unsigned short* ao  = ws + off; off += (size_t)BB * TT * HDH;  // total 48 MB

    const int nX = BB * TT * EE;       // 4M
    const int nW = HH * DH * EE;       // 1M

    cvt_bf16_kernel<<<nX / 256, 256, 0, stream>>>(x, xb, nX);
    transpose_w_kernel<<<nW / 256, 256, 0, stream>>>(Wq, WqT);
    transpose_w_kernel<<<nW / 256, 256, 0, stream>>>(Wk, WkT);
    transpose_w_kernel<<<nW / 256, 256, 0, stream>>>(Wv, WvT);
    cvt_bf16_kernel<<<nW / 256, 256, 0, stream>>>(Wo, Wob, nW);

    // 4096 waves each
    qkv_kernel<<<512, 256, 0, stream>>>(xb, WqT, WkT, WvT, qb, kb, vTb);
    attn_kernel<<<4096 / ATTN_WAVES, ATTN_WAVES * 32, 0, stream>>>(qb, kb, vTb, ao);
    outproj_kernel<<<512, 256, 0, stream>>>(ao, Wob, bo, out);
}
